// DotProductAttention_6322191860075
// MI455X (gfx1250) — matrix operand order repaired and checked by reading, not compiled, not run on hardware
//
#include <hip/hip_runtime.h>

// Flash attention, B=16, SQ=SK=2048, D=256, fp32 I/O, f16 WMMA + fp32 accum.
// 8 waves/block; each wave owns 16 query rows. Key loop: 64 keys/tile with
// online softmax. K staged row-major f16, V staged transposed f16 in LDS.

typedef __attribute__((ext_vector_type(16))) _Float16 v16h;
typedef __attribute__((ext_vector_type(8)))  _Float16 v8h;
typedef __attribute__((ext_vector_type(2)))  _Float16 v2h;
typedef __attribute__((ext_vector_type(8)))  float    v8f;
typedef __attribute__((ext_vector_type(4)))  float    v4f;

#define WAVES 8
#define BQ 16           // query rows per wave
#define ROWS_PER_BLOCK 128
#define KT 64           // keys per tile
#define BDIM 256

__device__ __forceinline__ float redmax16(float x) {
  x = fmaxf(x, __shfl_xor(x, 1, 32));
  x = fmaxf(x, __shfl_xor(x, 2, 32));
  x = fmaxf(x, __shfl_xor(x, 4, 32));
  x = fmaxf(x, __shfl_xor(x, 8, 32));
  return x;
}
__device__ __forceinline__ float redsum16(float x) {
  x += __shfl_xor(x, 1, 32);
  x += __shfl_xor(x, 2, 32);
  x += __shfl_xor(x, 4, 32);
  x += __shfl_xor(x, 8, 32);
  return x;
}

__global__ __launch_bounds__(BDIM) void fa_kernel(
    const float* __restrict__ Q, const float* __restrict__ K,
    const float* __restrict__ V, const int* __restrict__ VL,
    float* __restrict__ out)
{
  constexpr int SQc = 2048, SKc = 2048, Dc = 256;

  __shared__ _Float16 kt[KT * Dc];        // [key][d]   32 KB
  __shared__ _Float16 vt[Dc * KT];        // [d][key]   32 KB (transposed)
  __shared__ _Float16 pt[WAVES][BQ * KT]; // per-wave P 16 KB

  const int tid  = threadIdx.x;
  const int wave = tid >> 5;
  const int lane = tid & 31;
  const int m    = lane & 15;   // fragment M-row / B-fragment N-col
  const int hi   = lane >> 4;   // lane half selector

  const int blocksPerB = SQc / ROWS_PER_BLOCK;               // 16
  const int b     = blockIdx.x / blocksPerB;
  const int qbase = (blockIdx.x % blocksPerB) * ROWS_PER_BLOCK;
  const int q0    = qbase + wave * BQ;

  const int vlen = VL[b];

  const float* Qb = Q + (size_t)b * SQc * Dc;
  const float* Kb = K + (size_t)b * SKc * Dc;
  const float* Vb = V + (size_t)b * SKc * Dc;

  // ---- Q tile in registers, directly in A-fragment layout (8 k-chunks).
  v16h qf[8];
#pragma unroll
  for (int c = 0; c < 8; ++c) {
    const float* p0 = Qb + (size_t)(q0 + m) * Dc + c * 32 + hi * 16;
    v4f a0 = *(const v4f*)(p0);
    v4f a1 = *(const v4f*)(p0 + 4);
    v4f a2 = *(const v4f*)(p0 + 8);
    v4f a3 = *(const v4f*)(p0 + 12);
    v16h af;
#pragma unroll
    for (int j = 0; j < 4; ++j) {
      af[j]      = (_Float16)a0[j];
      af[4 + j]  = (_Float16)a1[j];
      af[8 + j]  = (_Float16)a2[j];
      af[12 + j] = (_Float16)a3[j];
    }
    qf[c] = af;
  }

  // O accumulators: 16 dim-tiles x 8 f32 (C layout: VGPR i = rows i / i+8).
  float o_acc[16][8];
#pragma unroll
  for (int t = 0; t < 16; ++t)
#pragma unroll
    for (int i = 0; i < 8; ++i) o_acc[t][i] = 0.0f;

  float m_s[8], l_s[8];
#pragma unroll
  for (int i = 0; i < 8; ++i) { m_s[i] = -1e30f; l_s[i] = 0.0f; }

  const int nKT = SKc / KT; // 32
  for (int ktile = 0; ktile < nKT; ++ktile) {
    const int kbase = ktile * KT;

    // ---- K tile: fp32 -> f16, row-major, packed 16B LDS stores.
    // 64 rows x 32 groups of 8 floats = 2048 items, 8 per thread.
#pragma unroll
    for (int r = 0; r < 8; ++r) {
      int id  = tid + r * BDIM;
      int key = id >> 5;
      int d8  = id & 31;
      const float* src = Kb + (size_t)(kbase + key) * Dc + d8 * 8;
      v4f k0 = *(const v4f*)(src);
      v4f k1 = *(const v4f*)(src + 4);
      v8h h;
#pragma unroll
      for (int j = 0; j < 4; ++j) {
        h[j]     = (_Float16)k0[j];
        h[4 + j] = (_Float16)k1[j];
      }
      *(v8h*)(&kt[key * Dc + d8 * 8]) = h;
    }

    // ---- V tile: fp32 -> f16, transposed [d][key], key-pair packed b32
    // stores (consecutive lanes -> consecutive pairs: conflict-free).
#pragma unroll
    for (int r = 0; r < 8; ++r) {
      int id = tid + r * BDIM;
      int p  = id & 31;         // key pair 0..31
      int d4 = id >> 5;         // float4 column 0..63
      int k0 = 2 * p;
      v4f va = *(const v4f*)(Vb + (size_t)(kbase + k0)     * Dc + d4 * 4);
      v4f vb = *(const v4f*)(Vb + (size_t)(kbase + k0 + 1) * Dc + d4 * 4);
#pragma unroll
      for (int j = 0; j < 4; ++j) {
        v2h h; h[0] = (_Float16)va[j]; h[1] = (_Float16)vb[j];
        *(v2h*)(&vt[(d4 * 4 + j) * KT + k0]) = h;
      }
    }
    __syncthreads();

    // Prefetch next tile while computing this one.
    if (ktile + 1 < nKT) {
      __builtin_prefetch(Kb + (size_t)(kbase + KT) * Dc + tid * 64, 0, 1);
      __builtin_prefetch(Vb + (size_t)(kbase + KT) * Dc + tid * 64, 0, 1);
    }

    // ---- S = Q*K^T: 4 key N-tiles x 8 k-chunks = 32 WMMAs.
    v8f s[4];
#pragma unroll
    for (int nt = 0; nt < 4; ++nt) s[nt] = (v8f){};
#pragma unroll
    for (int c = 0; c < 8; ++c) {
#pragma unroll
      for (int nt = 0; nt < 4; ++nt) {
        v16h bk = *(const v16h*)(&kt[(nt * 16 + m) * Dc + c * 32 + hi * 16]);
        s[nt] = __builtin_amdgcn_wmma_f32_16x16x32_f16(
            false, qf[c], false, bk, (short)0, s[nt], false, false);
      }
    }

    // ---- scale, mask, online softmax (row = i + hi*8 per accum register).
    bool ok[4];
#pragma unroll
    for (int nt = 0; nt < 4; ++nt) ok[nt] = (kbase + nt * 16 + m) < vlen;
#pragma unroll
    for (int i = 0; i < 8; ++i) {
      float x[4];
#pragma unroll
      for (int nt = 0; nt < 4; ++nt)
        x[nt] = ok[nt] ? s[nt][i] * 0.0625f : -10000.0f;
      float tmax = fmaxf(fmaxf(x[0], x[1]), fmaxf(x[2], x[3]));
      tmax = redmax16(tmax);
      float mnew  = fmaxf(m_s[i], tmax);
      float alpha = __expf(m_s[i] - mnew);
      float psum  = 0.0f;
#pragma unroll
      for (int nt = 0; nt < 4; ++nt) {
        float p = __expf(x[nt] - mnew);
        psum += p;
        pt[wave][(i + hi * 8) * KT + nt * 16 + m] = (_Float16)p;
      }
      l_s[i] = l_s[i] * alpha + redsum16(psum);
      m_s[i] = mnew;
#pragma unroll
      for (int t = 0; t < 16; ++t) o_acc[t][i] *= alpha;
    }
    asm volatile("s_wait_dscnt 0" ::: "memory");

    // ---- Reload P as two 16x32 A fragments.
    v16h pf[2];
#pragma unroll
    for (int c = 0; c < 2; ++c) {
      v8h plo = *(const v8h*)(&pt[wave][m * KT + c * 32 + hi * 16]);
      v8h phi = *(const v8h*)(&pt[wave][m * KT + c * 32 + hi * 16 + 8]);
      v16h af;
#pragma unroll
      for (int j = 0; j < 8; ++j) { af[j] = plo[j]; af[8 + j] = phi[j]; }
      pf[c] = af;
    }

    // ---- O += P*V: 16 dim-tiles x 2 k-chunks = 32 WMMAs.
#pragma unroll
    for (int t = 0; t < 16; ++t) {
      v8f c;
#pragma unroll
      for (int i = 0; i < 8; ++i) c[i] = o_acc[t][i];
#pragma unroll
      for (int kc = 0; kc < 2; ++kc) {
        v16h bv = *(const v16h*)(&vt[(t * 16 + m) * KT + kc * 32 + hi * 16]);
        c = __builtin_amdgcn_wmma_f32_16x16x32_f16(
            false, pf[kc], false, bv, (short)0, c, false, false);
      }
#pragma unroll
      for (int i = 0; i < 8; ++i) o_acc[t][i] = c[i];
    }
    __syncthreads();
  }

  // ---- Epilogue: normalize and store fp32 output.
#pragma unroll
  for (int i = 0; i < 8; ++i) {
    float inv = 1.0f / l_s[i];
    int row = q0 + i + hi * 8;
    float* op = out + (size_t)(b * SQc + row) * Dc;
#pragma unroll
    for (int t = 0; t < 16; ++t) op[t * 16 + m] = o_acc[t][i] * inv;
  }
}

extern "C" void kernel_launch(void* const* d_in, const int* in_sizes, int n_in,
                              void* d_out, int out_size, void* d_ws, size_t ws_size,
                              hipStream_t stream) {
  (void)in_sizes; (void)n_in; (void)out_size; (void)d_ws; (void)ws_size;
  const float* Q  = (const float*)d_in[0];
  const float* K  = (const float*)d_in[1];
  const float* V  = (const float*)d_in[2];
  const int*   VL = (const int*)d_in[3];
  float* out = (float*)d_out;

  const int B = 16, SQ = 2048;
  dim3 grid(B * (SQ / ROWS_PER_BLOCK));  // 256 blocks
  dim3 block(BDIM);                      // 8 waves
  fa_kernel<<<grid, block, 0, stream>>>(Q, K, V, VL, out);
}
